// NeuralCoilLayer_32057635897559
// MI455X (gfx1250) — compile-verified, matrix operand order checked by reading, not run on hardware
//
#include <hip/hip_runtime.h>
#include <hip/hip_bf16.h>

typedef __attribute__((ext_vector_type(16))) __bf16 v16bf;
typedef __attribute__((ext_vector_type(8)))  __bf16 v8bf;
typedef __attribute__((ext_vector_type(8)))  float  v8f;

#define NB   32      // batch
#define SL   64      // seq len
#define NF   48      // features
#define GD   49      // n+1 groups
#define KC   2352    // 48*49 contraction size (flattened as c = g*48 + k)
#define KT   74      // ceil(2352/32) K-tiles
#define MROWS 2304   // 48*48 output rows
#define MT   144     // M tiles of 16

// workspace offsets (bytes)
#define A_BYTES   10911744UL            // 144*74*32*16 bf16 elems * 2
#define T_BYTES   294912UL              // 32*2304 f32
#define S_BYTES   6144UL                // 32*48 f32

// LDS layout (bytes)
#define BS_BYTES   151552               // 74*2*32*16 bf16
#define WSM_OFF    BS_BYTES             // 32*49 f32 = 6272
#define SRED_OFF   (WSM_OFF + 6272)     // 6*32 f32 = 768
#define WL_OFF     (SRED_OFF + 768)     // 48 f32 = 192
#define SMEM_BYTES (WL_OFF + 192)       // 158784 < 160KB

// ---------------------------------------------------------------------------
// One-time: swizzle interaction into per-wave WMMA A fragments (bf16), with
// flattened contraction index c = g*48 + k  (so per-step B building is
// vectorizable: 8-aligned runs of c share g and have consecutive k).
// I element (m, k, g) lives at I[m*2352 + k*49 + g].
// Fragment layout: Abuf[((mtile*74 + kt)*32 + lane)*16 + e]
// ISA 16-bit A 16x32 layout: element e -> Koff = (e&8?16:0)+(lane>>4)*8+(e&7).
// ---------------------------------------------------------------------------
__global__ void prep_abuf_kernel(const float* __restrict__ I,
                                 __bf16* __restrict__ A) {
    long idx = (long)blockIdx.x * blockDim.x + threadIdx.x;
    if (idx >= 5455872L) return;
    int  e    = (int)(idx & 15);
    int  ln   = (int)((idx >> 4) & 31);
    long tile = idx >> 9;                 // mtile*74 + kt
    int  kt   = (int)(tile % 74);
    int  mtile= (int)(tile / 74);
    int  hi   = ln >> 4;
    int  koff = ((e & 8) ? 16 : 0) + hi * 8 + (e & 7);
    int  c    = kt * 32 + koff;           // c = g*48 + k
    int  m    = mtile * 16 + (ln & 15);
    float v = 0.0f;
    if (c < KC) {
        int g = c / NF, k = c - g * NF;
        v = I[(long)m * KC + k * GD + g];
    }
    A[idx] = (__bf16)v;
}

__global__ void zero_f32_kernel(float* p, int n) {
    int i = blockIdx.x * blockDim.x + threadIdx.x;
    if (i < n) p[i] = 0.0f;
}

// ---------------------------------------------------------------------------
// One step of the scan. grid = 48 (one WG per output row-block i),
// block = 192 (6 waves: wave = wt*2 + nt; wt in 0..2 -> j tile, nt -> batch half)
// ---------------------------------------------------------------------------
__global__ __launch_bounds__(192)
void step_kernel(const float* __restrict__ x,      // [32][64][48]
                 const float* __restrict__ Watt,   // [48]
                 const __bf16* __restrict__ Abuf,
                 const float* __restrict__ Tprev,  // [32][48][48]
                 float*       __restrict__ Tcur,
                 const float* __restrict__ Sprev,  // [32][48]
                 float*       __restrict__ Scur,
                 float*       __restrict__ out,    // [32][64][48]
                 int t) {
    extern __shared__ char smem[];
    __bf16* Bs   = (__bf16*)smem;                  // [kt][nt][lane][16]
    float*  wsm  = (float*)(smem + WSM_OFF);       // scores -> softmax weights [32][49]
    float*  sRed = (float*)(smem + SRED_OFF);      // [6 waves][32 lanes]
    float*  Wl   = (float*)(smem + WL_OFF);        // W_att copy [48]

    const int i    = blockIdx.x;       // output row-block (the "i" index)
    const int tid  = threadIdx.x;
    const int lane = tid & 31;
    const int wave = tid >> 5;

    if (tid < NF) Wl[tid] = Watt[tid];

    // ---- Phase 0: emit out[t-1] (softmax over i of 48*s), WGs 0..31 ----
    if (t > 0 && i < NB && tid < NF) {
        const int b = i;
        float v = Sprev[b * NF + tid] * (float)NF;
        float mx = -3.0e38f;
        for (int j = 0; j < NF; ++j) mx = fmaxf(mx, Sprev[b * NF + j] * (float)NF);
        float sum = 0.0f;
        for (int j = 0; j < NF; ++j) sum += __expf(Sprev[b * NF + j] * (float)NF - mx);
        out[((long)b * SL + (t - 1)) * NF + tid] = __expf(v - mx) / sum;
    }
    __syncthreads();

    // ---- Phase 1a: attention scores silu(ns . W_att), all (b,g) pairs ----
    for (int p = tid; p < NB * GD; p += 192) {
        int b = p / GD, g = p - b * GD;
        const float* row = (g == 0) ? (x + ((long)b * SL + t) * NF)
                                    : (Tprev + (long)b * MROWS + (g - 1) * NF);
        float acc = 0.0f;
        #pragma unroll 8
        for (int k = 0; k < NF; ++k) acc += row[k] * Wl[k];
        wsm[p] = acc / (1.0f + __expf(-acc));      // SiLU
    }
    __syncthreads();

    // ---- Phase 1b: softmax over g per batch (32 threads) ----
    if (tid < NB) {
        const int b = tid;
        float mx = -3.0e38f;
        for (int g = 0; g < GD; ++g) mx = fmaxf(mx, wsm[b * GD + g]);
        float sum = 0.0f;
        for (int g = 0; g < GD; ++g) sum += __expf(wsm[b * GD + g] - mx);
        float inv = 1.0f / sum;
        for (int g = 0; g < GD; ++g) wsm[b * GD + g] = __expf(wsm[b * GD + g] - mx) * inv;
    }
    __syncthreads();

    // ---- Phase 1c: build swizzled bf16 B fragments (wns) in LDS ----
    // With c = g*48 + k, every 8-aligned run of 8 fragment elements has a
    // single g and consecutive k -> two float4 loads + one 16B ds store.
    // runs: [kt][nt][lane][half], total 74*2*32*2 = 9472
    for (int p = tid; p < KT * 2 * 32 * 2; p += 192) {
        int half = p & 1;                  // e<8 vs e>=8
        int ln   = (p >> 1) & 31;
        int nt   = (p >> 6) & 1;
        int kt   = p >> 7;
        int b  = nt * 16 + (ln & 15);
        int hi = ln >> 4;
        int c0 = kt * 32 + half * 16 + hi * 8;   // 8-aligned, run stays in one g
        v8bf vals;
        if (c0 < KC) {
            int g  = c0 / NF;
            int k0 = c0 - g * NF;
            const float* row = (g == 0)
                ? (x + ((long)b * SL + t) * NF + k0)
                : (Tprev + (long)b * MROWS + (g - 1) * NF + k0);
            float w = wsm[b * GD + g];
            float4 r0 = *(const float4*)row;
            float4 r1 = *(const float4*)(row + 4);
            vals[0] = (__bf16)(w * r0.x); vals[1] = (__bf16)(w * r0.y);
            vals[2] = (__bf16)(w * r0.z); vals[3] = (__bf16)(w * r0.w);
            vals[4] = (__bf16)(w * r1.x); vals[5] = (__bf16)(w * r1.y);
            vals[6] = (__bf16)(w * r1.z); vals[7] = (__bf16)(w * r1.w);
        } else {
            vals = (v8bf)(__bf16)0.0f;
        }
        *(v8bf*)(Bs + (((kt * 2 + nt) * 32 + ln) << 4) + half * 8) = vals;
    }
    __syncthreads();

    // ---- Phase 2: WMMA GEMM  T[b, i*48+j] = Mb . wns ----
    const int wt = wave >> 1;          // j tile within row-block (0..2)
    const int nt = wave & 1;           // batch half
    const int mtile = i * 3 + wt;
    v8f acc = {0.f, 0.f, 0.f, 0.f, 0.f, 0.f, 0.f, 0.f};
    const __bf16* Ab = Abuf + ((long)mtile * KT) * 512 + lane * 16;
    const __bf16* Bb = Bs + nt * 512 + lane * 16;
    for (int kt = 0; kt < KT; ++kt) {
        __builtin_prefetch(Ab + (kt + 6) * 512, 0, 1);   // global_prefetch
        v16bf a  = *(const v16bf*)(Ab + kt * 512);
        v16bf bb = *(const v16bf*)(Bb + kt * 1024);
        acc = __builtin_amdgcn_wmma_f32_16x16x32_bf16(
                  false, a, false, bb, (short)0, acc, false, false);
    }

    // ---- Epilogue: scatter T, partial s = T[b,i,:] . x_t[b,:] ----
    // D layout: lane L, vgpr r -> M = r + (L>=16?8:0), N = L&15
    const int b     = nt * 16 + (lane & 15);
    const int jbase = wt * 16 + ((lane & 16) ? 8 : 0);
    const float* xtb = x + ((long)b * SL + t) * NF;
    float sp = 0.0f;
    #pragma unroll
    for (int r = 0; r < 8; ++r) {
        int j = jbase + r;
        float v = acc[r];
        Tcur[(long)b * MROWS + i * NF + j] = v;
        sp += v * xtb[j];
    }
    sRed[wave * 32 + lane] = sp;
    __syncthreads();

    // deterministic fixed-order reduction of s over the 3 j-tiles
    if (tid < NB) {
        int b2 = tid, nt2 = b2 >> 4, l0 = b2 & 15;
        float s = 0.0f;
        #pragma unroll
        for (int w2 = 0; w2 < 3; ++w2) {
            int wv = w2 * 2 + nt2;
            s += sRed[wv * 32 + l0] + sRed[wv * 32 + l0 + 16];
        }
        Scur[b2 * NF + i] = s;
    }
}

// ---------------------------------------------------------------------------
// After step 63: out for t=63 + copy T_final into d_out tail.
// ---------------------------------------------------------------------------
__global__ void finalize_kernel(const float* __restrict__ S1,
                                const float* __restrict__ T1,
                                float* __restrict__ out) {
    const int b = blockIdx.x;
    if (threadIdx.x < NF) {
        float v = S1[b * NF + threadIdx.x] * (float)NF;
        float mx = -3.0e38f;
        for (int j = 0; j < NF; ++j) mx = fmaxf(mx, S1[b * NF + j] * (float)NF);
        float sum = 0.0f;
        for (int j = 0; j < NF; ++j) sum += __expf(S1[b * NF + j] * (float)NF - mx);
        out[((long)b * SL + (SL - 1)) * NF + threadIdx.x] = __expf(v - mx) / sum;
    }
    for (int p = threadIdx.x; p < MROWS; p += blockDim.x)
        out[NB * SL * NF + (long)b * MROWS + p] = T1[(long)b * MROWS + p];
}

extern "C" void kernel_launch(void* const* d_in, const int* in_sizes, int n_in,
                              void* d_out, int out_size, void* d_ws, size_t ws_size,
                              hipStream_t stream) {
    const float* x    = (const float*)d_in[0];   // [32,64,48]
    const float* Watt = (const float*)d_in[1];   // [48]
    const float* I    = (const float*)d_in[2];   // [48,48,48,49]
    float* out = (float*)d_out;
    char*  ws  = (char*)d_ws;

    __bf16* Abuf = (__bf16*)ws;
    float* T0 = (float*)(ws + A_BYTES);
    float* T1 = (float*)(ws + A_BYTES + T_BYTES);
    float* S0 = (float*)(ws + A_BYTES + 2 * T_BYTES);
    float* S1 = (float*)(ws + A_BYTES + 2 * T_BYTES + S_BYTES);
    float* Tb[2] = {T0, T1};
    float* Sb[2] = {S0, S1};

    // one-time (per call, deterministic): bf16-swizzle interaction; zero T_prev(-1)
    prep_abuf_kernel<<<(5455872 + 255) / 256, 256, 0, stream>>>(I, Abuf);
    zero_f32_kernel<<<(NB * MROWS + 255) / 256, 256, 0, stream>>>(T1, NB * MROWS);

    for (int t = 0; t < SL; ++t) {
        int cur = t & 1, prev = cur ^ 1;
        step_kernel<<<NF, 192, SMEM_BYTES, stream>>>(
            x, Watt, Abuf, Tb[prev], Tb[cur], Sb[prev], Sb[cur], out, t);
    }
    finalize_kernel<<<NB, 64, 0, stream>>>(Sb[1], Tb[1], out);
}
